// MoELoRADelta_26431228740192
// MI455X (gfx1250) — compile-verified, matrix operand order checked by reading, not run on hardware
//
#include <hip/hip_runtime.h>
#include <hip/hip_bf16.h>

typedef __attribute__((ext_vector_type(16))) _Float16 v16h;
typedef __attribute__((ext_vector_type(8)))  float    v8f;
typedef __attribute__((ext_vector_type(4)))  float    v4f;

#define D_IN   3840
#define D_OUT  3840
#define NEXP   4
#define RANK   16
#define NTILE1 5             // 4 expert mid tiles + 1 router tile
#define KCH1   (D_IN / 32)   // 120 K-chunks for GEMM1 (f16 K=32)
#define OT2    (D_OUT / 16)  // 240 output N-tiles for GEMM2
#define WAVES  8
#define SCALE  4.0f          // alpha / sqrt(r) = 16/4

// K index within a 32-wide f16 WMMA K-chunk for (lane, half):
// lanes 0-15: halves 0..7 -> K 0..7,  halves 8..15 -> K 16..23
// lanes 16-31: halves 0..7 -> K 8..15, halves 8..15 -> K 24..31
__device__ __forceinline__ int kpat(int lane, int h) {
    return h + ((h < 8) ? 0 : 8) + ((lane >> 4) << 3);
}

// Pre-swizzle W1 (cols 0..63 = A[e][r][:], cols 64..67 = router_w, 68..79 = 0)
// into per-(chunk, tile, lane) v16h B-operand layout.
__global__ void prep_w1(const float* __restrict__ Aw, const float* __restrict__ Rw,
                        _Float16* __restrict__ w1) {
    int idx = blockIdx.x * blockDim.x + threadIdx.x;
    int h    = idx & 15;
    int lane = (idx >> 4) & 31;
    int rest = idx >> 9;          // c * NTILE1 + t
    int t    = rest % NTILE1;
    int c    = rest / NTILE1;
    if (c >= KCH1) return;
    int k = c * 32 + kpat(lane, h);
    int n = t * 16 + (lane & 15);
    float v = 0.0f;
    if (n < NEXP * RANK) {
        int e = n >> 4, r = n & 15;
        v = Aw[(size_t)(e * RANK + r) * D_IN + k];
    } else if (n < NEXP * RANK + NEXP) {
        v = Rw[(size_t)(n - NEXP * RANK) * D_IN + k];
    }
    w1[idx] = (_Float16)v;
}

// Pre-swizzle W2[k=e*16+r][o] = B[e][o][r] into per-(otile, kchunk, lane) layout.
__global__ void prep_w2(const float* __restrict__ Bw, _Float16* __restrict__ w2) {
    int idx = blockIdx.x * blockDim.x + threadIdx.x;
    int h    = idx & 15;
    int lane = (idx >> 4) & 31;
    int rest = idx >> 9;          // ot * 2 + kc
    int kc   = rest & 1;
    int ot   = rest >> 1;
    if (ot >= OT2) return;
    int k = kc * 32 + kpat(lane, h);
    int e = k >> 4, r = k & 15;
    int o = ot * 16 + (lane & 15);
    w2[idx] = (_Float16)Bw[(size_t)(e * D_OUT + o) * RANK + r];
}

__global__ __launch_bounds__(WAVES * 32)
void moe_lora_kernel(const float* __restrict__ x,
                     const v16h* __restrict__ w1,
                     const v16h* __restrict__ w2,
                     float* __restrict__ out) {
    const int lane = threadIdx.x & 31;
    const int wave = threadIdx.x >> 5;
    const int grp  = lane >> 4;      // lane group 0/1
    const int col  = lane & 15;
    const int t0   = blockIdx.x * 16;  // 16 tokens per block

    __shared__ __align__(16) float smem[16 * 64 + 64 + 64];
    float* mid_s   = smem;            // [16][64] unscaled mid (fp32 partial sums)
    float* logit_s = smem + 1024;     // [16][4]
    float* gate_s  = smem + 1088;     // [16][4] gate * 4.0

    for (int i = threadIdx.x; i < 1088; i += blockDim.x) smem[i] = 0.0f;
    __syncthreads();

    // ---------------- GEMM1: partial mid + router logits per wave ------------
    v8f acc[NTILE1];
#pragma unroll
    for (int t = 0; t < NTILE1; ++t) acc[t] = (v8f){};

    const float* xrow = x + (size_t)(t0 + col) * D_IN;

    for (int c = wave; c < KCH1; c += WAVES) {
        int base = c * 32 + grp * 8;
        v4f x0 = *(const v4f*)(xrow + base);
        v4f x1 = *(const v4f*)(xrow + base + 4);
        v4f x2 = *(const v4f*)(xrow + base + 16);
        v4f x3 = *(const v4f*)(xrow + base + 20);
        v16h a;
#pragma unroll
        for (int q = 0; q < 4; ++q) {
            a[q]      = (_Float16)x0[q];
            a[q + 4]  = (_Float16)x1[q];
            a[q + 8]  = (_Float16)x2[q];
            a[q + 12] = (_Float16)x3[q];
        }
#pragma unroll
        for (int t = 0; t < NTILE1; ++t) {
            v16h b = w1[(size_t)(c * NTILE1 + t) * 32 + lane];
            acc[t] = __builtin_amdgcn_wmma_f32_16x16x32_f16(
                false, a, false, b, (short)0, acc[t], false, false);
        }
    }

    // Reduce partials into LDS (C-layout: lane col = n, vgpr i -> m = i + grp*8)
#pragma unroll
    for (int t = 0; t < NEXP; ++t) {
#pragma unroll
        for (int i = 0; i < 8; ++i) {
            int m = i + grp * 8;
            atomicAdd(&mid_s[m * 64 + t * 16 + col], acc[t][i]);
        }
    }
    if (col < NEXP) {
#pragma unroll
        for (int i = 0; i < 8; ++i) {
            int m = i + grp * 8;
            atomicAdd(&logit_s[m * 4 + col], acc[4][i]);
        }
    }
    __syncthreads();

    // ---------------- Top-2 sparse softmax gates (lanes 0..15 of wave 0) -----
    if (threadIdx.x < 16) {
        float l[4];
#pragma unroll
        for (int e = 0; e < 4; ++e) l[e] = logit_s[threadIdx.x * 4 + e];
        int i0 = 0; float m0 = l[0];
#pragma unroll
        for (int e = 1; e < 4; ++e) if (l[e] > m0) { m0 = l[e]; i0 = e; }
        int i1 = -1; float m1 = -3.4e38f;
#pragma unroll
        for (int e = 0; e < 4; ++e) if (e != i0 && l[e] > m1) { m1 = l[e]; i1 = e; }
        float e1  = __expf(m1 - m0);
        float inv = SCALE / (1.0f + e1);
#pragma unroll
        for (int e = 0; e < 4; ++e) gate_s[threadIdx.x * 4 + e] = 0.0f;
        gate_s[threadIdx.x * 4 + i0] = inv;
        gate_s[threadIdx.x * 4 + i1] = e1 * inv;
    }
    __syncthreads();

    // ---------------- Build gate-scaled f16 A operand for GEMM2 (K=64) -------
    v16h a2[2];
#pragma unroll
    for (int kc = 0; kc < 2; ++kc) {
        int kb = kc * 32 + grp * 8;
        const float* row = mid_s + col * 64;
        v4f m0v = *(const v4f*)(row + kb);
        v4f m1v = *(const v4f*)(row + kb + 4);
        v4f m2v = *(const v4f*)(row + kb + 16);
        v4f m3v = *(const v4f*)(row + kb + 20);
        float g0 = gate_s[col * 4 + (kb >> 4)];
        float g1 = gate_s[col * 4 + ((kb + 16) >> 4)];
#pragma unroll
        for (int q = 0; q < 4; ++q) {
            a2[kc][q]      = (_Float16)(m0v[q] * g0);
            a2[kc][q + 4]  = (_Float16)(m1v[q] * g0);
            a2[kc][q + 8]  = (_Float16)(m2v[q] * g1);
            a2[kc][q + 12] = (_Float16)(m3v[q] * g1);
        }
    }

    // ---------------- GEMM2: delta tiles, waves split the 240 N-tiles --------
    for (int ot = wave; ot < OT2; ot += WAVES) {
        v16h b0 = w2[(size_t)(ot * 2 + 0) * 32 + lane];
        v16h b1 = w2[(size_t)(ot * 2 + 1) * 32 + lane];
        v8f o = (v8f){};
        o = __builtin_amdgcn_wmma_f32_16x16x32_f16(false, a2[0], false, b0,
                                                   (short)0, o, false, false);
        o = __builtin_amdgcn_wmma_f32_16x16x32_f16(false, a2[1], false, b1,
                                                   (short)0, o, false, false);
#pragma unroll
        for (int i = 0; i < 8; ++i) {
            out[(size_t)(t0 + i + grp * 8) * D_OUT + ot * 16 + col] = o[i];
        }
    }
}

extern "C" void kernel_launch(void* const* d_in, const int* in_sizes, int n_in,
                              void* d_out, int out_size, void* d_ws, size_t ws_size,
                              hipStream_t stream) {
    const float* x  = (const float*)d_in[0];   // [T, 3840]
    const float* rw = (const float*)d_in[1];   // [4, 3840]
    const float* Aw = (const float*)d_in[2];   // [4, 16, 3840]
    const float* Bw = (const float*)d_in[3];   // [4, 3840, 16]
    float* out = (float*)d_out;

    _Float16* w1 = (_Float16*)d_ws;
    const size_t W1N = (size_t)KCH1 * NTILE1 * 32 * 16;  // 307200 halves
    const size_t W2N = (size_t)OT2 * 2 * 32 * 16;        // 245760 halves
    _Float16* w2 = w1 + W1N;                             // 32B-aligned offset

    int T = in_sizes[0] / D_IN;  // 8192 tokens

    prep_w1<<<(int)((W1N + 255) / 256), 256, 0, stream>>>(Aw, rw, w1);
    prep_w2<<<(int)((W2N + 255) / 256), 256, 0, stream>>>(Bw, w2);
    moe_lora_kernel<<<T / 16, WAVES * 32, 0, stream>>>(
        x, (const v16h*)w1, (const v16h*)w2, out);
}
